// ClassicalSelfAttention_65481071405130
// MI455X (gfx1250) — matrix operand order repaired and checked by reading, not compiled, not run on hardware
//
#include <hip/hip_runtime.h>

typedef __bf16 bhalf;
typedef __attribute__((ext_vector_type(16))) __bf16 v16bf;
typedef __attribute__((ext_vector_type(8)))  __bf16 v8bf;
typedef __attribute__((ext_vector_type(8)))  float  v8f;
typedef __attribute__((ext_vector_type(4)))  unsigned int u32x4;
typedef __attribute__((ext_vector_type(8)))  int i32x8;
typedef __attribute__((ext_vector_type(4)))  int i32x4;

#define EDIM 1024
#define MTOT 8192
#define SLEN 2048

#if __has_builtin(__builtin_amdgcn_tensor_load_to_lds) && __has_builtin(__builtin_amdgcn_s_wait_tensorcnt)
#define HAVE_TDM 1
#else
#define HAVE_TDM 0
#endif

#if __has_builtin(__builtin_amdgcn_global_load_async_to_lds_b128)
#define HAVE_ASYNC 1
#else
#define HAVE_ASYNC 0
#endif

// ---------------------------------------------------------------------------
// helpers
// ---------------------------------------------------------------------------
static __device__ inline v8f zero8() {
  v8f z;
#pragma unroll
  for (int i = 0; i < 8; ++i) z[i] = 0.0f;
  return z;
}

static __device__ inline v16bf pack16(v8bf lo, v8bf hi) {
  v16bf r;
#pragma unroll
  for (int i = 0; i < 8; ++i) { r[i] = lo[i]; r[8 + i] = hi[i]; }
  return r;
}

static __device__ inline v8bf cvt8(v8f x) {
  v8bf r;
#pragma unroll
  for (int i = 0; i < 8; ++i) r[i] = (__bf16)x[i];
  return r;
}

// A fragment (16x32, bf16): lane<16 holds row lane, k = kb..kb+7 and 16+kb..16+kb+7
// with kb = (lane>>4)*8.  Two 16-byte contiguous loads per lane.
static __device__ inline v16bf load_frag_a_bf(const bhalf* base, int ld, int lane) {
  const int r  = lane & 15;
  const int kb = (lane >> 4) * 8;
  const bhalf* p = base + (size_t)r * ld + kb;
  v8bf lo = *(const v8bf*)(p);
  v8bf hi = *(const v8bf*)(p + 16);
  return pack16(lo, hi);
}

static __device__ inline v16bf load_frag_a_f32(const float* base, int ld, int lane) {
  const int r  = lane & 15;
  const int kb = (lane >> 4) * 8;
  const float* p = base + (size_t)r * ld + kb;
  v8f lo = *(const v8f*)(p);
  v8f hi = *(const v8f*)(p + 16);
  return pack16(cvt8(lo), cvt8(hi));
}

static __device__ inline v16bf load_frag_a(const bhalf* b, int ld, int lane) { return load_frag_a_bf(b, ld, lane); }
static __device__ inline v16bf load_frag_a(const float* b, int ld, int lane) { return load_frag_a_f32(b, ld, lane); }

// B fragment (32x16): B[k][n] taken from src[n][k] (row n contiguous in k).
static __device__ inline v16bf load_frag_b_bf(const bhalf* base, int ld, int lane) {
  const int n   = lane & 15;
  const int kb2 = (lane >> 4) * 8;
  const bhalf* p = base + (size_t)n * ld + kb2;
  v8bf lo = *(const v8bf*)(p);
  v8bf hi = *(const v8bf*)(p + 16);
  return pack16(lo, hi);
}

static __device__ inline v16bf load_frag_b_f32(const float* base, int ld, int lane) {
  const int n   = lane & 15;
  const int kb2 = (lane >> 4) * 8;
  const float* p = base + (size_t)n * ld + kb2;
  v8f lo = *(const v8f*)(p);
  v8f hi = *(const v8f*)(p + 16);
  return pack16(cvt8(lo), cvt8(hi));
}

static __device__ inline v8f wmma_bf16(v16bf a, v16bf b, v8f c) {
  return __builtin_amdgcn_wmma_f32_16x16x32_bf16(false, a, false, b, (short)0, c,
                                                 false, false);
}

static __device__ inline unsigned lds_addr_of(const void* p) {
  return (unsigned)(unsigned long long)p;  // low 32 bits = LDS byte offset
}

// ---------------------------------------------------------------------------
// Generic GEMM  D[M=8192, N] = A[8192,1024] * W[N,1024]^T  (bf16 WMMA, f32 acc)
// W tile (128 rows x 32 k, f32) staged per block via the Tensor Data Mover.
// A tile (128 rows x 32 k) staged via async global->LDS copies when available.
// MODE 0: qkv -> o0=q(*0.125), o1=k, o2=v^T ; MODE 1: of=f32, o0=bf16
// MODE 2: of = sigmoid(val) * gsrc
// ---------------------------------------------------------------------------
template <int MODE, typename TA>
__global__ void __launch_bounds__(128)
gemm_k(const TA* __restrict__ A, const float* __restrict__ W,
       bhalf* __restrict__ o0, bhalf* __restrict__ o1, bhalf* __restrict__ o2,
       float* __restrict__ of, const float* __restrict__ gsrc) {
  constexpr int NTOT = (MODE == 0) ? 3072 : 1024;
  __shared__ __attribute__((aligned(128))) float Wt[128 * 32];   // 16 KiB
#if HAVE_ASYNC
  __shared__ __attribute__((aligned(128))) TA At[128 * 32];      // <= 16 KiB
#endif

  const int tid  = threadIdx.x;
  const int lane = tid & 31;
  const int wave = tid >> 5;
  const int rowblk = blockIdx.y * 128;
  const int colblk = blockIdx.x * 128;
  const int row0 = rowblk + (wave >> 1) * 64;
  const int col0 = colblk + (wave & 1) * 64;
  const int hi = lane >> 4;
  const int ln = lane & 15;

  v8f acc[4][4];
#pragma unroll
  for (int i = 0; i < 4; ++i)
#pragma unroll
    for (int j = 0; j < 4; ++j) acc[i][j] = zero8();

  for (int k0 = 0; k0 < EDIM; k0 += 32) {
    __syncthreads();  // previous-iteration LDS reads complete

#if HAVE_TDM
    if (wave == 0) {
      // ---- TDM: stage W[colblk..+127][k0..+31] (f32) into Wt ----
      const unsigned long long ga =
          (unsigned long long)(const void*)(W + (size_t)colblk * EDIM + k0);
      u32x4 g0;
      g0[0] = 1u;                                   // count=1, user descriptor
      g0[1] = lds_addr_of(Wt);                      // lds_addr
      g0[2] = (unsigned)ga;                         // global_addr[31:0]
      g0[3] = ((unsigned)(ga >> 32) & 0x01FFFFFFu) | 0x80000000u;  // addr[56:32] | type=2
      i32x8 g1;
      g1[0] = (int)(2u << 16);                      // data_size = 4 bytes
      g1[1] = (int)(1024u << 16);                   // tensor_dim0 = 1024 (K)
      g1[2] = (int)((unsigned)(NTOT & 0xFFFF) << 16);  // tensor_dim1 lo16
      g1[3] = (int)(32u << 16);                     // tile_dim0 = 32 (tensor_dim1 hi16 = 0)
      g1[4] = 128;                                  // tile_dim1 = 128, tile_dim2 = 0
      g1[5] = 1024;                                 // tensor_dim0_stride
      g1[6] = 0;
      g1[7] = 0;
      i32x4 gz; gz[0] = 0; gz[1] = 0; gz[2] = 0; gz[3] = 0;
#if __clang_major__ >= 23
      i32x8 gz8;
#pragma unroll
      for (int z = 0; z < 8; ++z) gz8[z] = 0;
      __builtin_amdgcn_tensor_load_to_lds(g0, g1, gz, gz, gz8, 0);
#else
      __builtin_amdgcn_tensor_load_to_lds(g0, g1, gz, gz, 0);
#endif
      __builtin_amdgcn_s_wait_tensorcnt(0);
    }
#else
    // fallback: cooperative plain-load staging of the W tile
    {
      const float* sp = W + (size_t)(colblk + tid) * EDIM + k0;
      float* dp = Wt + tid * 32;
#pragma unroll
      for (int u = 0; u < 8; ++u)
        *(i32x4*)(dp + u * 4) = *(const i32x4*)(sp + u * 4);
    }
#endif

#if HAVE_ASYNC
    // ---- async copy A[rowblk..+127][k0..+31] into At (one row per thread) ----
    {
      const TA* sp = A + (size_t)(rowblk + tid) * EDIM + k0;
      TA* dp = At + tid * 32;
      constexpr int NB  = (int)(32 * sizeof(TA) / 16);  // 16B chunks per row
      constexpr int EPB = (int)(16 / sizeof(TA));       // elements per 16B
#pragma unroll
      for (int u = 0; u < NB; ++u)
        __builtin_amdgcn_global_load_async_to_lds_b128(
            (__attribute__((address_space(1))) i32x4*)(void*)(sp + u * EPB),
            (__attribute__((address_space(3))) i32x4*)(void*)(dp + u * EPB),
            0, 0);
#if __has_builtin(__builtin_amdgcn_s_wait_asynccnt)
      __builtin_amdgcn_s_wait_asynccnt(0);
#else
      asm volatile("s_wait_asynccnt 0x0" ::: "memory");
#endif
    }
#endif
    __syncthreads();  // tiles visible to all waves

    v16bf af[4], bfr[4];
#pragma unroll
    for (int i = 0; i < 4; ++i) {
#if HAVE_ASYNC
      af[i] = load_frag_a(At + ((wave >> 1) * 64 + 16 * i) * 32, 32, lane);
#else
      af[i] = load_frag_a(A + (size_t)(row0 + 16 * i) * EDIM + k0, EDIM, lane);
#endif
    }
#pragma unroll
    for (int j = 0; j < 4; ++j)
      bfr[j] = load_frag_b_f32(Wt + ((wave & 1) * 64 + 16 * j) * 32, 32, lane);
#pragma unroll
    for (int i = 0; i < 4; ++i)
#pragma unroll
      for (int j = 0; j < 4; ++j) acc[i][j] = wmma_bf16(af[i], bfr[j], acc[i][j]);
  }

#pragma unroll
  for (int i = 0; i < 4; ++i) {
#pragma unroll
    for (int j = 0; j < 4; ++j) {
      const int colg = col0 + 16 * j + ln;
#pragma unroll
      for (int r = 0; r < 8; ++r) {
        const int rowg = row0 + 16 * i + hi * 8 + r;
        const float v = acc[i][j][r];
        if (MODE == 0) {
          const int sel = colg >> 10;
          const int cc = colg & 1023;
          if (sel == 0)
            o0[(size_t)rowg * EDIM + cc] = (bhalf)(v * 0.125f);
          else if (sel == 1)
            o1[(size_t)rowg * EDIM + cc] = (bhalf)v;
          else
            o2[(size_t)cc * MTOT + rowg] = (bhalf)v;  // v stored transposed
        } else if (MODE == 1) {
          of[(size_t)rowg * EDIM + colg] = v;
          o0[(size_t)rowg * EDIM + colg] = (bhalf)v;
        } else {
          const float g = 1.0f / (1.0f + __expf(-v));
          of[(size_t)rowg * EDIM + colg] = g * gsrc[(size_t)rowg * EDIM + colg];
        }
      }
    }
  }
}

// ---------------------------------------------------------------------------
// Flash-style attention: one block (4 waves) per 16-row query tile.
// ---------------------------------------------------------------------------
__global__ void __launch_bounds__(128)
attn_kernel(const bhalf* __restrict__ qb, const bhalf* __restrict__ kb,
            const bhalf* __restrict__ vT, bhalf* __restrict__ ob) {
  __shared__ __attribute__((aligned(16))) bhalf Pl[16 * 64];
  __shared__ float rmax[4][16];
  __shared__ float rsum[4][16];
  __shared__ float mrun[16];
  __shared__ float lrun[16];
  __shared__ float sfac[16];
  __shared__ float mnew[16];

  const int tid = threadIdx.x;
  const int wave = tid >> 5;
  const int lane = tid & 31;
  const int hi = lane >> 4;
  const int ln = lane & 15;

  const int q0 = blockIdx.x * 16;
  const int batch = q0 / SLEN;
  const int jbeg = batch * SLEN;
  const int jend = jbeg + SLEN;

  if (tid < 16) { mrun[tid] = -3.0e38f; lrun[tid] = 0.0f; }
  __syncthreads();

  v8f acc[16];
#pragma unroll
  for (int t = 0; t < 16; ++t) acc[t] = zero8();

  const bhalf* qbase = qb + (size_t)q0 * EDIM;

  for (int j = jbeg; j < jend; j += 64) {
    // ---- score tile: S_w[16x16] = q_tile * k[j+16w .. +15]^T over K=1024 ----
    v8f s = zero8();
    const bhalf* kbase = kb + (size_t)(j + wave * 16) * EDIM;
#pragma unroll 4
    for (int e = 0; e < EDIM; e += 32) {
      v16bf a = load_frag_a_bf(qbase + e, EDIM, lane);
      v16bf b = load_frag_b_bf(kbase + e, EDIM, lane);
      s = wmma_bf16(a, b, s);
    }

    // ---- per-row max (reduce over the 16 lanes of each half) ----
    float rm[8];
#pragma unroll
    for (int r = 0; r < 8; ++r) rm[r] = s[r];
#pragma unroll
    for (int mask = 1; mask < 16; mask <<= 1)
#pragma unroll
      for (int r = 0; r < 8; ++r) rm[r] = fmaxf(rm[r], __shfl_xor(rm[r], mask, 32));
    if (ln == 0)
#pragma unroll
      for (int r = 0; r < 8; ++r) rmax[wave][hi * 8 + r] = rm[r];
    __syncthreads();

    // ---- combine running max / compute rescale factor ----
    if (tid < 16) {
      float m = mrun[tid];
#pragma unroll
      for (int w = 0; w < 4; ++w) m = fmaxf(m, rmax[w][tid]);
      const float sc = __expf(mrun[tid] - m);
      sfac[tid] = sc;
      mnew[tid] = m;
      mrun[tid] = m;
      lrun[tid] *= sc;
    }
    __syncthreads();

    // ---- P = exp(S - m), row sums, stage P in LDS, rescale accumulators ----
    float mr[8], sf[8], rs[8];
#pragma unroll
    for (int r = 0; r < 8; ++r) { mr[r] = mnew[hi * 8 + r]; sf[r] = sfac[hi * 8 + r]; }
#pragma unroll
    for (int r = 0; r < 8; ++r) {
      const float p = __expf(s[r] - mr[r]);
      rs[r] = p;
      Pl[(hi * 8 + r) * 64 + wave * 16 + ln] = (bhalf)p;
    }
#pragma unroll
    for (int mask = 1; mask < 16; mask <<= 1)
#pragma unroll
      for (int r = 0; r < 8; ++r) rs[r] += __shfl_xor(rs[r], mask, 32);
    if (ln == 0)
#pragma unroll
      for (int r = 0; r < 8; ++r) rsum[wave][hi * 8 + r] = rs[r];
#pragma unroll
    for (int t = 0; t < 16; ++t)
#pragma unroll
      for (int r = 0; r < 8; ++r) acc[t][r] *= sf[r];
    __syncthreads();

    if (tid < 16) {
      float l = lrun[tid];
#pragma unroll
      for (int w = 0; w < 4; ++w) l += rsum[w][tid];
      lrun[tid] = l;
    }

    // ---- attn @ V on this wave's 256-column slice ----
    const v16bf p0 = load_frag_a_bf(Pl, 64, lane);       // k = 0..31
    const v16bf p1 = load_frag_a_bf(Pl + 32, 64, lane);  // k = 32..63
    const bhalf* vbase = vT + (size_t)(wave * 256) * MTOT + j;
#pragma unroll
    for (int t = 0; t < 16; ++t) {
      const bhalf* vb = vbase + (size_t)(t * 16) * MTOT;
      v16bf b0 = load_frag_b_bf(vb, MTOT, lane);
      v16bf b1 = load_frag_b_bf(vb + 32, MTOT, lane);
      acc[t] = wmma_bf16(p0, b0, acc[t]);
      acc[t] = wmma_bf16(p1, b1, acc[t]);
    }
    __syncthreads();  // Pl / rmax / lrun consistent before next step
  }

  // ---- normalize and store ----
  float linv[8];
#pragma unroll
  for (int r = 0; r < 8; ++r) linv[r] = 1.0f / lrun[hi * 8 + r];
#pragma unroll
  for (int t = 0; t < 16; ++t) {
    const int colg = wave * 256 + t * 16 + ln;
#pragma unroll
    for (int r = 0; r < 8; ++r) {
      const int rowg = q0 + hi * 8 + r;
      ob[(size_t)rowg * EDIM + colg] = (bhalf)(acc[t][r] * linv[r]);
    }
  }
}

// ---------------------------------------------------------------------------
// launch
// ---------------------------------------------------------------------------
extern "C" void kernel_launch(void* const* d_in, const int* in_sizes, int n_in,
                              void* d_out, int out_size, void* d_ws, size_t ws_size,
                              hipStream_t stream) {
  const float* rot = (const float*)d_in[0];  // (3E, E)
  const float* ent = (const float*)d_in[1];  // (E, E)
  const float* x   = (const float*)d_in[2];  // (B, S, E)
  const float* gw  = (const float*)d_in[3];  // (E, E)
  float* out = (float*)d_out;                // (B, S, E) f32

  const size_t ME = (size_t)MTOT * EDIM;
  bhalf* q_bf  = (bhalf*)d_ws;     // 16 MiB each
  bhalf* k_bf  = q_bf + ME;
  bhalf* vT_bf = k_bf + ME;
  bhalf* at_bf = vT_bf + ME;
  bhalf* o_bf  = at_bf + ME;
  float* o_f   = (float*)(o_bf + ME);

  dim3 blk(128);
  // 1) QKV projection (N = 3072)
  gemm_k<0, float><<<dim3(24, 64), blk, 0, stream>>>(x, rot, q_bf, k_bf, vT_bf,
                                                     nullptr, nullptr);
  // 2) attention (512 query tiles of 16 rows)
  attn_kernel<<<dim3(512), blk, 0, stream>>>(q_bf, k_bf, vT_bf, at_bf);
  // 3) output projection (N = 1024)
  gemm_k<1, bhalf><<<dim3(8, 64), blk, 0, stream>>>(at_bf, ent, o_bf, nullptr,
                                                    nullptr, o_f, nullptr);
  // 4) gate projection + sigmoid * out (N = 1024)
  gemm_k<2, bhalf><<<dim3(8, 64), blk, 0, stream>>>(o_bf, gw, nullptr, nullptr,
                                                    nullptr, out, o_f);
}
